// GaussianMMD_57080115364694
// MI455X (gfx1250) — compile-verified, matrix operand order checked by reading, not run on hardware
//
#include <hip/hip_runtime.h>

// ---------------------------------------------------------------------------
// Gaussian MMD, N=8192, D=1024, sigma=2048.
// bf16 hi/lo split GEMM on v_wmma_f32_16x16x32_bf16, async global->LDS
// staging (ASYNCcnt), double-buffered LDS, fused exp+sum epilogue, symmetry
// for xx/yy, fp64 atomics for the scalar reduction.
// ---------------------------------------------------------------------------

#define MMD_N 8192
#define MMD_D 1024
#define MMD_SIGMA 2048.0f
#define PLANE (128 * 40)          // ushorts per LDS plane (row stride 40 = 80B)

typedef __attribute__((ext_vector_type(16))) __bf16 v16bf;
typedef __attribute__((ext_vector_type(8)))  float  v8f;

// ---------------- helpers ----------------

static __device__ inline unsigned short f32_to_bf16_rne(float f) {
    unsigned u = __float_as_uint(f);
    unsigned r = (u + 0x7FFFu + ((u >> 16) & 1u)) >> 16;
    return (unsigned short)r;
}
static __device__ inline float bf16_to_f32(unsigned short h) {
    return __uint_as_float(((unsigned)h) << 16);
}

static __device__ inline v8f wmma_bf16(v16bf a, v16bf b, v8f c) {
    // (neg_a, A, neg_b, B, c_mod, C, reuse_a, reuse_b)
    return __builtin_amdgcn_wmma_f32_16x16x32_bf16(false, a, false, b,
                                                   (short)0, c, false, false);
}

// Byte offset of a generic __shared__ pointer within LDS (AS3 ptrs are LDS offsets).
static __device__ inline unsigned lds_off(const void* p) {
    return (unsigned)(unsigned long long)(__attribute__((address_space(3))) const void*)p;
}

// Per-lane async DMA: 16 bytes global -> LDS, tracked by ASYNCcnt, no VGPR data.
static __device__ inline void async_b128(const void* gaddr, unsigned ldsoff) {
    asm volatile("global_load_async_to_lds_b128 %0, %1, off"
                 :: "v"(ldsoff), "v"((unsigned long long)gaddr)
                 : "memory");
}
static __device__ inline void wait_async0() {
    asm volatile("s_wait_asynccnt 0x0" ::: "memory");
}

// Load one 16x32 bf16 fragment from an LDS panel with row stride 40 ushorts
// (80 bytes: 16B aligned, conflict free). ISA layout for 16-bit A/B frags:
// lanes 0-15 hold K=0..7 (q0) / K=16..23 (q1) of row M=lane; lanes 16-31
// hold K=8..15 / K=24..31 (+16B).
static __device__ inline v16bf load_frag(const unsigned short* lds, int row, int laneHi) {
    union { v16bf v; uint4 q[2]; } f;
    const uint4* p = (const uint4*)(lds + row * 40 + laneHi * 8);
    f.q[0] = p[0];   // bytes [0,16)
    f.q[1] = p[2];   // bytes [32,48)
    return f.v;
}

// ---------------- kernels ----------------

__global__ void mmd_init_accum(double* accum) {
    if (threadIdx.x < 3) accum[threadIdx.x] = 0.0;
}

// Split fp32 -> bf16 hi + bf16 lo (residual), 4 elements / thread.
__global__ void mmd_prep(const float* __restrict__ src,
                         unsigned short* __restrict__ hi,
                         unsigned short* __restrict__ lo) {
    int t = blockIdx.x * blockDim.x + threadIdx.x;      // one float4
    float4 v = ((const float4*)src)[t];
    unsigned short h0 = f32_to_bf16_rne(v.x);
    unsigned short h1 = f32_to_bf16_rne(v.y);
    unsigned short h2 = f32_to_bf16_rne(v.z);
    unsigned short h3 = f32_to_bf16_rne(v.w);
    unsigned short l0 = f32_to_bf16_rne(v.x - bf16_to_f32(h0));
    unsigned short l1 = f32_to_bf16_rne(v.y - bf16_to_f32(h1));
    unsigned short l2 = f32_to_bf16_rne(v.z - bf16_to_f32(h2));
    unsigned short l3 = f32_to_bf16_rne(v.w - bf16_to_f32(h3));
    uint2 hv, lv;
    hv.x = (unsigned)h0 | ((unsigned)h1 << 16);
    hv.y = (unsigned)h2 | ((unsigned)h3 << 16);
    lv.x = (unsigned)l0 | ((unsigned)l1 << 16);
    lv.y = (unsigned)l2 | ((unsigned)l3 << 16);
    ((uint2*)hi)[t] = hv;
    ((uint2*)lo)[t] = lv;
}

// One wave per row: squared L2 norms in fp32.
__global__ void mmd_norms(const float* __restrict__ x, const float* __restrict__ y,
                          float* __restrict__ xn, float* __restrict__ yn) {
    int wave = threadIdx.x >> 5;
    int lane = threadIdx.x & 31;
    int row = blockIdx.x * 8 + wave;
    const float* src = blockIdx.y ? y : x;
    float*       dst = blockIdx.y ? yn : xn;
    const float* r = src + (size_t)row * MMD_D;
    float s = 0.0f;
    #pragma unroll 4
    for (int e = lane; e < MMD_D; e += 32) { float v = r[e]; s += v * v; }
    #pragma unroll
    for (int o = 16; o > 0; o >>= 1) s += __shfl_xor(s, o);
    if (lane == 0) dst[row] = s;
}

// Block: 128 (M) x 128 (N) tile; 8 waves as 4x2, each wave a 32x64 micro-tile
// (2x4 WMMA accumulators, 24 wmma / 24 ds_load_b128 per K-step).
// z: 0=xx, 1=yy, 2=xy.  Double-buffered LDS filled by async global->LDS DMA.
__global__ __launch_bounds__(256)
void mmd_main(const unsigned short* __restrict__ xhi, const unsigned short* __restrict__ xlo,
              const unsigned short* __restrict__ yhi, const unsigned short* __restrict__ ylo,
              const float* __restrict__ xn, const float* __restrict__ yn,
              double* __restrict__ accum) {
    const int z = blockIdx.z;
    const unsigned short *Agh, *Agl, *Bgh, *Bgl;
    const float *anG, *bnG;
    if (z == 0)      { Agh = xhi; Agl = xlo; Bgh = xhi; Bgl = xlo; anG = xn; bnG = xn; }
    else if (z == 1) { Agh = yhi; Agl = ylo; Bgh = yhi; Bgl = ylo; anG = yn; bnG = yn; }
    else             { Agh = xhi; Agl = xlo; Bgh = yhi; Bgl = ylo; anG = xn; bnG = yn; }

    const int iBase = blockIdx.y * 128;   // rows (i)
    const int jBase = blockIdx.x * 128;   // cols (j)
    if (z < 2 && (jBase + 128) <= iBase) return;   // fully below diagonal

    extern __shared__ char smem[];
    unsigned short* AhS = (unsigned short*)smem;         // [2][PLANE]
    unsigned short* AlS = AhS + 2 * PLANE;
    unsigned short* BhS = AlS + 2 * PLANE;
    unsigned short* BlS = BhS + 2 * PLANE;
    float* anS  = (float*)(BlS + 2 * PLANE);             // [128]
    float* bnS  = anS + 128;                             // [128]
    float* redS = bnS + 128;                             // [8]

    const int tid    = threadIdx.x;
    const int lane   = tid & 31;
    const int wave   = tid >> 5;
    const int lrow   = lane & 15;
    const int laneHi = (lane >> 4) & 1;
    const int waveM  = wave >> 1;   // 0..3 -> 32-row strips
    const int waveN  = wave & 1;    // 0..1 -> 64-col strips

    if (tid < 128) {
        anS[tid] = anG[iBase + tid];
        bnS[tid] = bnG[jBase + tid];
    }

    // Issue one 32-wide K panel (A 128x32 hi/lo + B 128x32 hi/lo) into buffer
    // `buf` via async DMA: 8 x b128 per thread, 32 KB per block per K-step.
    auto issue = [&](int kt2, int buf) {
        const int k0 = kt2 * 32;
        const int bofs = buf * PLANE;
        #pragma unroll
        for (int c = 0; c < 2; ++c) {
            int idx = tid + c * 256;
            int row = idx >> 2, col = idx & 3;          // col: 16B chunk in 32-elem row
            size_t gA = (size_t)(iBase + row) * MMD_D + k0 + col * 8;
            size_t gB = (size_t)(jBase + row) * MMD_D + k0 + col * 8;
            unsigned lofs = (unsigned)(row * 40 + col * 8);
            async_b128(Agh + gA, lds_off(AhS + bofs + lofs));
            async_b128(Agl + gA, lds_off(AlS + bofs + lofs));
            async_b128(Bgh + gB, lds_off(BhS + bofs + lofs));
            async_b128(Bgl + gB, lds_off(BlS + bofs + lofs));
        }
    };

    v8f zero = {};
    v8f acc[2][4];
    #pragma unroll
    for (int tm = 0; tm < 2; ++tm)
        #pragma unroll
        for (int tn = 0; tn < 4; ++tn) acc[tm][tn] = zero;

    issue(0, 0);   // prologue

    for (int kt = 0; kt < MMD_D / 32; ++kt) {
        const int buf = kt & 1;
        wait_async0();        // own batch kt landed (in-order per wave)
        __syncthreads();      // everyone's buffer `buf` complete; prior reads done
        if (kt + 1 < MMD_D / 32) issue(kt + 1, buf ^ 1);   // overlap with compute

        const unsigned short* Ah = AhS + buf * PLANE;
        const unsigned short* Al = AlS + buf * PLANE;
        const unsigned short* Bh = BhS + buf * PLANE;
        const unsigned short* Bl = BlS + buf * PLANE;

        v16bf ah[2], al[2], bh[4], bl[4];
        #pragma unroll
        for (int tm = 0; tm < 2; ++tm) {
            int r = waveM * 32 + tm * 16 + lrow;
            ah[tm] = load_frag(Ah, r, laneHi);
            al[tm] = load_frag(Al, r, laneHi);
        }
        #pragma unroll
        for (int tn = 0; tn < 4; ++tn) {
            int r = waveN * 64 + tn * 16 + lrow;
            bh[tn] = load_frag(Bh, r, laneHi);
            bl[tn] = load_frag(Bl, r, laneHi);
        }

        #pragma unroll
        for (int tm = 0; tm < 2; ++tm)
            #pragma unroll
            for (int tn = 0; tn < 4; ++tn) {
                acc[tm][tn] = wmma_bf16(ah[tm], bh[tn], acc[tm][tn]); // hi*hi
                acc[tm][tn] = wmma_bf16(ah[tm], bl[tn], acc[tm][tn]); // hi*lo
                acc[tm][tn] = wmma_bf16(al[tm], bh[tn], acc[tm][tn]); // lo*hi
            }
    }

    // Fused epilogue: exp kernel + symmetry weights + reduction.
    const float invS = 1.0f / MMD_SIGMA;
    float lsum = 0.0f;
    #pragma unroll
    for (int tm = 0; tm < 2; ++tm) {
        #pragma unroll
        for (int tn = 0; tn < 4; ++tn) {
            int jl = waveN * 64 + tn * 16 + lrow;
            int j  = jBase + jl;
            float bn = bnS[jl];
            #pragma unroll
            for (int e = 0; e < 8; ++e) {
                int il = waveM * 32 + tm * 16 + 8 * laneHi + e; // C layout: M=e+8*hi
                int i  = iBase + il;
                float d = acc[tm][tn][e];
                float v = __expf((2.0f * d - anS[il] - bn) * invS);
                if (z < 2) v *= (j > i) ? 2.0f : ((j == i) ? 1.0f : 0.0f);
                lsum += v;
            }
        }
    }
    #pragma unroll
    for (int o = 16; o > 0; o >>= 1) lsum += __shfl_xor(lsum, o);
    if (lane == 0) redS[wave] = lsum;
    __syncthreads();
    if (tid == 0) {
        float b = 0.0f;
        #pragma unroll
        for (int w = 0; w < 8; ++w) b += redS[w];
        atomicAdd(&accum[z], (double)b);
    }
}

__global__ void mmd_finalize(const double* __restrict__ accum, float* __restrict__ out) {
    double inv = 1.0 / ((double)MMD_N * (double)MMD_N);
    out[0] = (float)((accum[0] + accum[1] - 2.0 * accum[2]) * inv);
}

// ---------------- launch ----------------

extern "C" void kernel_launch(void* const* d_in, const int* in_sizes, int n_in,
                              void* d_out, int out_size, void* d_ws, size_t ws_size,
                              hipStream_t stream) {
    (void)in_sizes; (void)n_in; (void)out_size; (void)ws_size;
    const float* x = (const float*)d_in[0];
    const float* y = (const float*)d_in[1];
    float* out = (float*)d_out;

    const size_t ND = (size_t)MMD_N * MMD_D;
    char* ws = (char*)d_ws;
    double*         accum = (double*)ws;                       // 3 doubles
    unsigned short* xhi = (unsigned short*)(ws + 256);
    unsigned short* xlo = xhi + ND;
    unsigned short* yhi = xlo + ND;
    unsigned short* ylo = yhi + ND;
    float* xn = (float*)(ylo + ND);
    float* yn = xn + MMD_N;

    mmd_init_accum<<<1, 32, 0, stream>>>(accum);

    mmd_prep<<<MMD_N * MMD_D / (256 * 4), 256, 0, stream>>>(x, xhi, xlo);
    mmd_prep<<<MMD_N * MMD_D / (256 * 4), 256, 0, stream>>>(y, yhi, ylo);

    mmd_norms<<<dim3(MMD_N / 8, 2), 256, 0, stream>>>(x, y, xn, yn);

    const size_t smemBytes = (size_t)(4 * 2 * PLANE) * 2 + (128 + 128 + 8) * 4;
    mmd_main<<<dim3(MMD_N / 128, MMD_N / 128, 3), 256, smemBytes, stream>>>(
        xhi, xlo, yhi, ylo, xn, yn, accum);

    mmd_finalize<<<1, 1, 0, stream>>>(accum, out);
}
